// LSTM_48584670052543
// MI455X (gfx1250) — matrix operand order repaired and checked by reading, not compiled, not run on hardware
//
#include <hip/hip_runtime.h>
#include <hip/hip_bf16.h>

// ---------------------------------------------------------------------------
// LSTM on MI455X (gfx1250): bf16 WMMA (16x16x32, fp32 accum).
// Persistent recurrent kernel. A operand staged into LDS by the Tensor Data
// Mover (tensor_load_to_lds, TENSORcnt), double-buffered so DMA overlaps
// WMMA compute. Device-wide barrier per time step.
// ---------------------------------------------------------------------------

#define B_   128
#define T_   512
#define DIN  512
#define H_   1024
#define G4   4096          // 4*H
#define NBLK 64            // persistent grid: one block per 16 h-columns

#define CW      64                  // chunk K-width (elements)
#define A_PAD   72                  // 64 + 8 bf16 row padding (144B stride)
#define A_ONE   (128 * A_PAD * 2)   // 18432 B per chunk buffer
#define NCHUNK  24                  // K = 1536 = 24 * 64

typedef __attribute__((ext_vector_type(16))) __bf16 v16bf;
typedef __attribute__((ext_vector_type(8)))  __bf16 v8bf;
typedef __attribute__((ext_vector_type(8)))  float  v8f;
typedef __attribute__((ext_vector_type(4)))  unsigned int v4u;
typedef __attribute__((ext_vector_type(8)))  int v8i;
typedef __attribute__((ext_vector_type(4)))  int v4i;

// ---- helpers ---------------------------------------------------------------

__device__ __forceinline__ __bf16 f2bf(float f) {
  union { float f; unsigned u; } a; a.f = f;
  unsigned r = a.u + 0x7FFFu + ((a.u >> 16) & 1u);      // round-to-nearest-even
  union { unsigned short s; __bf16 b; } o;
  o.s = (unsigned short)(r >> 16);
  return o.b;
}

__device__ __forceinline__ float sigm(float x) {
  return 1.0f / (1.0f + __expf(-x));
}
__device__ __forceinline__ float tanh_fast(float x) {
  return 2.0f / (1.0f + __expf(-2.0f * x)) - 1.0f;
}

// A fragment (16x32 bf16, M x K). lane<16: row=lane, K in {k0..k0+7, k0+16..k0+23}
// lane>=16: row=lane-16, K halves shifted by 8. Caller passes p = rowbase + k0 + hi*8.
__device__ __forceinline__ v16bf ldA(const __bf16* p) {
  union { v16bf v; v8bf h[2]; } u;
  u.h[0] = *(const v8bf*)(p);
  u.h[1] = *(const v8bf*)(p + 16);
  return u.v;
}

// B fragment (32x16 bf16, K x N) for C = A * W^T: column n of B = row n of W.
// lane<16: N=lane, K in {k0..k0+7, k0+16..k0+23} ; lane>=16: N=lane-16, K halves shifted by 8.
// Caller passes p = Wrow(n) + k0 + hi*8.
__device__ __forceinline__ v16bf ldB(const __bf16* p) {
  union { v16bf v; v8bf h[2]; } u;
  u.h[0] = *(const v8bf*)(p);
  u.h[1] = *(const v8bf*)(p + 16);
  return u.v;
}

#define WMMA_BF16(A, Bf, C) \
  __builtin_amdgcn_wmma_f32_16x16x32_bf16(false, (A), false, (Bf), (short)0, (C), false, false)

// ---- Tensor Data Mover: DMA a [128 rows x 64 cols] bf16 tile into LDS ------
// Hardware adds 16B padding after every 128B row -> LDS row stride 144B (A_PAD).
// D# per CDNA5 ISA 8.3/8.4: group0 = {count/lds/global/type}, group1 = dims.
__device__ __forceinline__ void tdm_load_chunk(const __bf16* gsrc, unsigned lds_off,
                                               unsigned ld_elems) {
  unsigned long long ga = (unsigned long long)(uintptr_t)gsrc;
  v4u g0;
  g0[0] = 1u;                                             // count=1, user mode
  g0[1] = lds_off;                                        // lds_addr (bytes)
  g0[2] = (unsigned)ga;                                   // global_addr[31:0]
  g0[3] = (unsigned)((ga >> 32) & 0x01FFFFFFu)            // global_addr[56:32]
          | 0x80000000u;                                  // type=2 ("image")
  v8i g1;
  g1[0] = (int)((1u << 16)                                // data_size: 2 bytes
              | (1u << 20)                                // pad_enable
              | (4u << 22)                                // pad_interval: 32 DW (128B)
              | (3u << 25));                              // pad_amount: 4 DW (16B)
  g1[1] = (int)((ld_elems & 0xFFFFu) << 16);              // tensor_dim0 lo16 (abar=0)
  g1[2] = (int)((ld_elems >> 16) | (128u << 16));         // tensor_dim0 hi | tensor_dim1 lo
  g1[3] = (int)(((unsigned)CW) << 16);                    // tensor_dim1 hi=0 | tile_dim0=64
  g1[4] = (int)128u;                                      // tile_dim1=128, tile_dim2=0
  g1[5] = (int)ld_elems;                                  // tensor_dim0_stride lo32
  g1[6] = 0;                                              // stride hi | dim1_stride lo (2D)
  g1[7] = 0;
  v4i z4 = {0, 0, 0, 0};                                  // groups 2/3 unused (2D)
  v8i z8 = {0, 0, 0, 0, 0, 0, 0, 0};
  __builtin_amdgcn_tensor_load_to_lds(g0, g1, z4, z4, z8, 0);
}

// agent-scope grid barrier (sense via generation counter)
__device__ __forceinline__ void grid_barrier(unsigned* cnt, unsigned* gen, unsigned nb) {
  __threadfence();            // release this block's h-writes to device scope
  __syncthreads();
  if (threadIdx.x == 0) {
    unsigned my = __hip_atomic_load(gen, __ATOMIC_RELAXED, __HIP_MEMORY_SCOPE_AGENT);
    unsigned t  = __hip_atomic_fetch_add(cnt, 1u, __ATOMIC_ACQ_REL, __HIP_MEMORY_SCOPE_AGENT);
    if (t == nb - 1u) {
      __hip_atomic_store(cnt, 0u, __ATOMIC_RELAXED, __HIP_MEMORY_SCOPE_AGENT);
      __hip_atomic_fetch_add(gen, 1u, __ATOMIC_RELEASE, __HIP_MEMORY_SCOPE_AGENT);
    } else {
      while (__hip_atomic_load(gen, __ATOMIC_ACQUIRE, __HIP_MEMORY_SCOPE_AGENT) == my) {
        __builtin_amdgcn_s_sleep(8);
      }
    }
  }
  __syncthreads();
  __threadfence();            // acquire: invalidate so all waves see fresh h
}

// ---- prep kernels ----------------------------------------------------------

// X [B,T,D] f32 -> Xbf [T,B,D] bf16   (exactly T*B*D threads)
__global__ void k_transpose_X(const float* __restrict__ X, __bf16* __restrict__ Xbf) {
  int i = blockIdx.x * blockDim.x + threadIdx.x;   // i = (t*B + b)*D + d
  int d  = i & (DIN - 1);
  int tb = i >> 9;
  int b  = tb & (B_ - 1);
  int t  = tb >> 7;
  Xbf[i] = f2bf(X[((size_t)b * T_ + t) * DIN + d]);
}

__global__ void k_f32_to_bf16(const float* __restrict__ src, __bf16* __restrict__ dst, int n) {
  int i = blockIdx.x * blockDim.x + threadIdx.x;
  if (i < n) dst[i] = f2bf(src[i]);
}

// bias = b_ih + b_hh ; zero h ping-pong buffers ; reset barrier
__global__ void k_init(const float* __restrict__ bih, const float* __restrict__ bhh,
                       float* __restrict__ bias, __bf16* __restrict__ h0,
                       __bf16* __restrict__ h1, unsigned* __restrict__ bar) {
  int i = blockIdx.x * blockDim.x + threadIdx.x;   // 131072 threads
  if (i < G4) bias[i] = bih[i] + bhh[i];
  h0[i] = f2bf(0.0f);
  h1[i] = f2bf(0.0f);
  if (i < 2) bar[i] = 0u;
}

// ---- persistent recurrent kernel -------------------------------------------
// grid = 64 blocks x 128 threads (4 waves). Block wg owns h-cols [wg*16, wg*16+16).
// Wave w computes gate w's z block [128 x 16] via bf16 WMMA over K = H (h) + DIN (x_t).
// A operand (h || x_t) is TDM-DMA'd into LDS in 24 chunks of [128 x 64] bf16,
// double-buffered; the 32KB gate buffer zs aliases the two staging buffers.

__global__ __launch_bounds__(128, 1)
void k_lstm(const __bf16* __restrict__ Xbf, const __bf16* __restrict__ Whh,
            const __bf16* __restrict__ Wih, const float* __restrict__ bias,
            __bf16* __restrict__ h0buf, __bf16* __restrict__ h1buf,
            unsigned* __restrict__ bar) {
  __shared__ __align__(16) char smem[2 * A_ONE];   // 36864 B: As0 | As1 (zs aliases)
  float* zs = (float*)smem;                        // [4][128][16] f32 (aliased)
  const unsigned lds_base = (unsigned)(uintptr_t)smem;  // LDS byte offset (addr[31:0])

  const int tid   = threadIdx.x;
  const int wave  = tid >> 5;                      // gate index 0..3
  const int lane  = tid & 31;
  const int rA    = lane & 15;
  const int hi    = lane >> 4;                     // 0/1 K-half select
  const int wg    = blockIdx.x;
  const int cbase = wg * 16;

  // gate-phase mapping: thread handles col gc, rows gb0 + 8*jj
  const int gc  = tid & 15;
  const int gb0 = tid >> 4;                        // 0..7
  const float bi = bias[          cbase + gc];
  const float bf = bias[1 * H_ +  cbase + gc];
  const float bg = bias[2 * H_ +  cbase + gc];
  const float bo = bias[3 * H_ +  cbase + gc];

  float creg[16];
  #pragma unroll
  for (int j = 0; j < 16; ++j) creg[j] = 0.0f;

  // this wave's 16 output columns -> rows of W (size 4096)
  const int wrow = wave * H_ + cbase + rA;
  const __bf16* whh_row = Whh + (size_t)wrow * H_  + hi * 8;
  const __bf16* wih_row = Wih + (size_t)wrow * DIN + hi * 8;

  unsigned* bcnt = bar;
  unsigned* bgen = bar + 1;

  for (int t = 0; t < T_; ++t) {
    const __bf16* hread  = (t & 1) ? h1buf : h0buf;
    __bf16*       hwrite = (t & 1) ? h0buf : h1buf;
    const __bf16* xt0 = Xbf + (size_t)t * (B_ * DIN);

    v8f acc[8];
    #pragma unroll
    for (int m = 0; m < 8; ++m) acc[m] = (v8f){0,0,0,0,0,0,0,0};

    // prologue: DMA chunk 0 into buffer 0 (wave 0 issues for the block)
    if (wave == 0) {
      tdm_load_chunk(hread, lds_base, H_);
    }

    // ---- K = 1536 in 24 chunks of 64 (16 from h, 8 from x_t) ----
    for (int ch = 0; ch < NCHUNK; ++ch) {
      const int nx = ch + 1;
      if (wave == 0) {
        if (nx < NCHUNK) {
          const __bf16* srcn = (nx < 16) ? (hread + nx * CW)
                                         : (xt0 + (nx - 16) * CW);
          unsigned ldn = (nx < 16) ? H_ : DIN;
          tdm_load_chunk(srcn, lds_base + (unsigned)(nx & 1) * A_ONE, ldn);
          __builtin_amdgcn_s_wait_tensorcnt((short)1);   // chunk ch landed
        } else {
          __builtin_amdgcn_s_wait_tensorcnt((short)0);   // drain
        }
      }
      __syncthreads();   // chunk ch visible to all waves

      const __bf16* wr = (ch < 16) ? (whh_row + ch * CW)
                                   : (wih_row + (ch - 16) * CW);
      const __bf16* Ab = (const __bf16*)(smem + (ch & 1) * A_ONE);
      #pragma unroll
      for (int kt = 0; kt < 2; ++kt) {
        v16bf bfrag = ldB(wr + kt * 32);
        #pragma unroll
        for (int m = 0; m < 8; ++m) {
          v16bf afrag = ldA(&Ab[(m * 16 + rA) * A_PAD + kt * 32 + hi * 8]);
          acc[m] = WMMA_BF16(afrag, bfrag, acc[m]);
        }
      }
      __syncthreads();   // readers done before this buffer is DMA-overwritten
    }

    // scatter z block to LDS: C layout: vgpr j -> row j + hi*8, col rA
    #pragma unroll
    for (int m = 0; m < 8; ++m) {
      #pragma unroll
      for (int j = 0; j < 8; ++j) {
        zs[((wave * 128 + m * 16 + j + hi * 8) << 4) + rA] = acc[m][j];
      }
    }
    __syncthreads();

    // fused gates: c = sig(f)*c + sig(i)*tanh(g); h = sig(o)*tanh(c)
    #pragma unroll
    for (int jj = 0; jj < 16; ++jj) {
      const int row  = gb0 + (jj << 3);
      const int base = (row << 4) + gc;
      float zi = zs[           base] + bi;
      float zf = zs[1 * 2048 + base] + bf;
      float zg = zs[2 * 2048 + base] + bg;
      float zo = zs[3 * 2048 + base] + bo;
      float cn = sigm(zf) * creg[jj] + sigm(zi) * tanh_fast(zg);
      creg[jj] = cn;
      float hn = sigm(zo) * tanh_fast(cn);
      hwrite[(size_t)row * H_ + cbase + gc] = f2bf(hn);
    }

    grid_barrier(bcnt, bgen, NBLK);
  }
}

// ---- final projection: out[128,512] = h_final @ W_out^T + b_out ------------
// grid = 8 blocks x 128 threads; wave handles 16 output cols, 8 M-tiles.
__global__ __launch_bounds__(128, 1)
void k_out(const __bf16* __restrict__ hfin, const __bf16* __restrict__ Wout,
           const float* __restrict__ bout, float* __restrict__ out) {
  const int tid  = threadIdx.x;
  const int wave = tid >> 5;
  const int lane = tid & 31;
  const int rA   = lane & 15;
  const int hi   = lane >> 4;
  const int n0   = blockIdx.x * 64 + wave * 16;

  const __bf16* wrow = Wout + (size_t)(n0 + rA) * H_ + hi * 8;
  const __bf16* hr   = hfin + hi * 8;

  v8f acc[8];
  #pragma unroll
  for (int m = 0; m < 8; ++m) acc[m] = (v8f){0,0,0,0,0,0,0,0};

  for (int k0 = 0; k0 < H_; k0 += 32) {
    v16bf bfrag = ldB(wrow + k0);
    #pragma unroll
    for (int m = 0; m < 8; ++m) {
      v16bf afrag = ldA(hr + (size_t)(m * 16 + rA) * H_ + k0);
      acc[m] = WMMA_BF16(afrag, bfrag, acc[m]);
    }
  }

  const int col = n0 + rA;
  const float bo = bout[col];
  #pragma unroll
  for (int m = 0; m < 8; ++m) {
    #pragma unroll
    for (int j = 0; j < 8; ++j) {
      out[(size_t)(m * 16 + j + hi * 8) * 512 + col] = acc[m][j] + bo;
    }
  }
}

// ---------------------------------------------------------------------------

extern "C" void kernel_launch(void* const* d_in, const int* in_sizes, int n_in,
                              void* d_out, int out_size, void* d_ws, size_t ws_size,
                              hipStream_t stream) {
  const float* X    = (const float*)d_in[0];   // [B,T,DIN]
  const float* Wih  = (const float*)d_in[1];   // [4H, DIN]
  const float* Whh  = (const float*)d_in[2];   // [4H, H]
  const float* bih  = (const float*)d_in[3];   // [4H]
  const float* bhh  = (const float*)d_in[4];   // [4H]
  const float* Wout = (const float*)d_in[5];   // [DOUT, H]
  const float* bout = (const float*)d_in[6];   // [DOUT]

  // workspace carve (256B aligned)
  char* ws = (char*)d_ws;
  size_t off = 0;
  auto carve = [&](size_t bytes) {
    char* p = ws + off;
    off += (bytes + 255) & ~(size_t)255;
    return p;
  };
  __bf16*   Xbf     = (__bf16*)carve((size_t)T_ * B_ * DIN * 2);  // 64 MB
  __bf16*   Whh_bf  = (__bf16*)carve((size_t)G4 * H_ * 2);        //  8 MB
  __bf16*   Wih_bf  = (__bf16*)carve((size_t)G4 * DIN * 2);       //  4 MB
  __bf16*   Wout_bf = (__bf16*)carve((size_t)512 * H_ * 2);       //  1 MB
  float*    bias    = (float*) carve((size_t)G4 * 4);
  __bf16*   h0      = (__bf16*)carve((size_t)B_ * H_ * 2);
  __bf16*   h1      = (__bf16*)carve((size_t)B_ * H_ * 2);
  unsigned* bar     = (unsigned*)carve(256);

  const int thr = 256;
  k_transpose_X<<<(T_ * B_ * DIN) / thr, thr, 0, stream>>>(X, Xbf);
  k_f32_to_bf16<<<(G4 * H_ + thr - 1) / thr, thr, 0, stream>>>(Whh, Whh_bf, G4 * H_);
  k_f32_to_bf16<<<(G4 * DIN + thr - 1) / thr, thr, 0, stream>>>(Wih, Wih_bf, G4 * DIN);
  k_f32_to_bf16<<<(512 * H_ + thr - 1) / thr, thr, 0, stream>>>(Wout, Wout_bf, 512 * H_);
  k_init<<<(B_ * H_) / thr, thr, 0, stream>>>(bih, bhh, bias, h0, h1, bar);

  k_lstm<<<NBLK, 128, 0, stream>>>(Xbf, Whh_bf, Wih_bf, bias, h0, h1, bar);

  // T=512 steps: final h was written to h0 (t=511 odd -> writes h0)
  k_out<<<8, 128, 0, stream>>>(h0, Wout_bf, bout, (float*)d_out);
}
